// self_attention_43336220016804
// MI455X (gfx1250) — matrix operand order repaired and checked by reading, not compiled, not run on hardware
//
#include <hip/hip_runtime.h>

typedef __bf16 bf16_t;
typedef bf16_t v16bf __attribute__((ext_vector_type(16)));
typedef bf16_t v8bf  __attribute__((ext_vector_type(8)));
typedef float  v8f   __attribute__((ext_vector_type(8)));

#define NB 4
#define CH 64
#define NN 4096

__device__ __forceinline__ bf16_t f2bf(float f) { return (bf16_t)f; }

__device__ __forceinline__ v8f wmma_bf16(v16bf a, v16bf b, v8f c) {
  return __builtin_amdgcn_wmma_f32_16x16x32_bf16(false, a, false, b, (short)0, c,
                                                 false, false);
}
// Variant with B-operand reuse hint: legal when the previous instruction is an
// identical WMMA sharing the B operand (we order the call sites to guarantee it).
__device__ __forceinline__ v8f wmma_bf16_rb(v16bf a, v16bf b, v8f c) {
  return __builtin_amdgcn_wmma_f32_16x16x32_bf16(false, a, false, b, (short)0, c,
                                                 false, true);
}

// A-operand (16x32 bf16). Layout: lane = M (row); lower half-wave holds
// K in {0..15}, upper half-wave holds K in {16..31} (matches B operands).
// `row` points at K=0 of this lane's row (>=32 contiguous bf16 K values).
__device__ __forceinline__ v16bf ldA_bf(const bf16_t* row, int hl) {
  const v8bf lo = *(const v8bf*)(row + hl * 16);       // K: hl*16 .. hl*16+7
  const v8bf hi = *(const v8bf*)(row + hl * 16 + 8);   // K: hl*16+8 .. +15
  v16bf a;
#pragma unroll
  for (int i = 0; i < 8; ++i) { a[i] = lo[i]; a[i + 8] = hi[i]; }
  return a;
}

// ---------------------------------------------------------------------------
// Kernel 0: one-time f32 -> bf16 conversion of the four 64x64 weight matrices.
// Removes hundreds of redundant per-wave v_cvt_pk_bf16_f32 ops from the hot
// kernels and halves weight fetch traffic.
// ---------------------------------------------------------------------------
__global__ __launch_bounds__(256) void wcvt_kernel(
    const float* __restrict__ w_theta, const float* __restrict__ w_phi,
    const float* __restrict__ w_g, const float* __restrict__ w_out,
    bf16_t* __restrict__ wb) {  // wb: 4 x [64][64] bf16, contiguous
  const int idx = blockIdx.x * 256 + threadIdx.x;   // 0..16383
  const int m = idx >> 12, e = idx & 4095;
  const float* src = (m == 0) ? w_theta : (m == 1) ? w_phi
                   : (m == 2) ? w_g     : w_out;
  wb[idx] = f2bf(src[e]);
}

// ---------------------------------------------------------------------------
// Kernel 1: 1x1-conv projections theta/phi/g via WMMA.
//   thetaT[b][n][c], phiT[b][n][c]  (token-major: contiguous channels)
//   gbuf[b][o][n]                   (channel-major: contiguous tokens)
// One wave per (batch, 16-token tile); 24 WMMAs per wave, xv0/xv1 B-operands
// shared across the three projection matrices (reuse_b on followers).
// ---------------------------------------------------------------------------
__global__ __launch_bounds__(256) void proj_kernel(
    const float* __restrict__ x,
    const bf16_t* __restrict__ wtb, const bf16_t* __restrict__ wpb,
    const bf16_t* __restrict__ wgb,
    const float* __restrict__ b_theta, const float* __restrict__ b_phi,
    const float* __restrict__ b_g,
    bf16_t* __restrict__ thetaT, bf16_t* __restrict__ phiT,
    bf16_t* __restrict__ gbuf) {
  __shared__ float xt[8][CH][17];  // padded to dodge bank conflicts

  const int wv   = threadIdx.x >> 5;
  const int lane = threadIdx.x & 31;
  const int wid  = blockIdx.x * 8 + wv;          // 0..1023
  const int b    = wid >> 8;                     // 256 tiles per batch
  const int n0   = (wid & 255) << 4;
  const int hl   = lane >> 4;
  const int ln   = lane & 15;

  // Cooperative load of the 64ch x 16tok x-tile (f32) into LDS.
  const float* xb = x + (size_t)b * CH * NN + n0;
#pragma unroll
  for (int it = 0; it < 8; ++it) {
    const int idx = it * 32 + lane;              // 0..255
    const int c = idx >> 2, q = (idx & 3) << 2;
    const float4 v = *(const float4*)(xb + (size_t)c * NN + q);
    xt[wv][c][q + 0] = v.x; xt[wv][c][q + 1] = v.y;
    xt[wv][c][q + 2] = v.z; xt[wv][c][q + 3] = v.w;
  }
  __syncthreads();

  // B-operands (32x16 bf16): lane = token column; lower half K=c0..c0+15,
  // upper half K=c0+16..c0+31.
  v16bf xv0, xv1;
#pragma unroll
  for (int r = 0; r < 16; ++r) {
    xv0[r] = f2bf(xt[wv][hl * 16 + r][ln]);        // c = 0..31
    xv1[r] = f2bf(xt[wv][32 + hl * 16 + r][ln]);   // c = 32..63
  }

#pragma unroll
  for (int j = 0; j < 4; ++j) {                    // output-channel 16-chunks
    const int obase = 16 * j + 8 * hl;
    const size_t wrow = (size_t)(16 * j + ln) * CH;

    v8f aT = {}, aP = {}, aG = {};
    // K-chunk 0 (c 0..31): three identical WMMAs sharing xv0 -> reuse_b.
    aT = wmma_bf16   (ldA_bf(wtb + wrow, hl), xv0, aT);
    aP = wmma_bf16_rb(ldA_bf(wpb + wrow, hl), xv0, aP);
    aG = wmma_bf16_rb(ldA_bf(wgb + wrow, hl), xv0, aG);
    // K-chunk 1 (c 32..63): share xv1.
    aT = wmma_bf16   (ldA_bf(wtb + wrow + 32, hl), xv1, aT);
    aP = wmma_bf16_rb(ldA_bf(wpb + wrow + 32, hl), xv1, aP);
    aG = wmma_bf16_rb(ldA_bf(wgb + wrow + 32, hl), xv1, aG);

    // theta / phi: token-major 16-byte stores of 8 consecutive channels.
    v8bf oT, oP;
#pragma unroll
    for (int r = 0; r < 8; ++r) {
      oT[r] = f2bf(aT[r] + b_theta[obase + r]);
      oP[r] = f2bf(aP[r] + b_phi[obase + r]);
    }
    *(v8bf*)(thetaT + (size_t)(b * NN + n0 + ln) * CH + obase) = oT;
    *(v8bf*)(phiT   + (size_t)(b * NN + n0 + ln) * CH + obase) = oP;
    // g: channel-major scatter.
#pragma unroll
    for (int r = 0; r < 8; ++r)
      gbuf[(size_t)(b * CH + obase + r) * NN + n0 + ln] =
          f2bf(aG[r] + b_g[obase + r]);
  }
}

// ---------------------------------------------------------------------------
// Kernel 2: fused flash-attention + output projection + ReLU + BN + residual.
// One wave per (batch, 16-query tile). S^T = phi^T * theta puts the
// softmax-over-keys reduction per-lane (8 VGPRs) + one shfl_xor(16); P^T's
// D-layout converts to the PV B-operand with one half-wave swap. 8 WMMAs per
// 32-key step x 128 steps, + 8 epilogue WMMAs, all with reuse-hint chains.
// ---------------------------------------------------------------------------
__global__ __launch_bounds__(256) void attn_kernel(
    const bf16_t* __restrict__ thetaT, const bf16_t* __restrict__ phiT,
    const bf16_t* __restrict__ gbuf, const bf16_t* __restrict__ wob,
    const float* __restrict__ b_out,
    const float* __restrict__ gamma, const float* __restrict__ beta,
    const float* __restrict__ x, float* __restrict__ out) {
  const int wv   = threadIdx.x >> 5;
  const int lane = threadIdx.x & 31;
  const int wid  = blockIdx.x * 8 + wv;
  const int b    = wid >> 8;
  const int n0   = (wid & 255) << 4;
  const int hl   = lane >> 4;
  const int ln   = lane & 15;

  // Persistent Q B-operand: queries n0..n0+15, K = channel.
  const bf16_t* qrow = thetaT + (size_t)(b * NN + n0 + ln) * CH;
  const v16bf qb0 = *(const v16bf*)(qrow + hl * 16);        // c 0..31
  const v16bf qb1 = *(const v16bf*)(qrow + 32 + hl * 16);   // c 32..63

  const bf16_t* phiB = phiT + (size_t)b * NN * CH;
  const bf16_t* gB   = gbuf + (size_t)b * CH * NN;

  float mrun = -3.0e38f, lsum = 0.0f;
  v8f y0 = {}, y1 = {}, y2 = {}, y3 = {};

  for (int m0 = 0; m0 < NN; m0 += 32) {
    // --- S^T tiles: D[M=key, N=query], two 16-key subtiles.
    // Ordered so consecutive WMMAs share the B operand (qb0 / qb1).
    v8f s0 = {}, s1 = {};
    const bf16_t* kr = phiB + (size_t)(m0 + ln) * CH;
    __builtin_prefetch(kr + 32 * CH, 0, 1);  // next m-step (L2-resident)
    s0 = wmma_bf16   (ldA_bf(kr, hl), qb0, s0);
    s1 = wmma_bf16_rb(ldA_bf(kr + (size_t)16 * CH, hl), qb0, s1);
    s0 = wmma_bf16   (ldA_bf(kr + 32, hl), qb1, s0);
    s1 = wmma_bf16_rb(ldA_bf(kr + (size_t)16 * CH + 32, hl), qb1, s1);

    // --- online softmax (per query = per lane column) ---
    float tmax = s0[0];
#pragma unroll
    for (int r = 1; r < 8; ++r) tmax = fmaxf(tmax, s0[r]);
#pragma unroll
    for (int r = 0; r < 8; ++r) tmax = fmaxf(tmax, s1[r]);
    tmax = fmaxf(tmax, __shfl_xor(tmax, 16));
    const float mnew  = fmaxf(mrun, tmax);
    const float alpha = __expf(mrun - mnew);
    mrun = mnew;

    float psum = 0.0f;
#pragma unroll
    for (int r = 0; r < 8; ++r) { s0[r] = __expf(s0[r] - mnew); psum += s0[r]; }
#pragma unroll
    for (int r = 0; r < 8; ++r) { s1[r] = __expf(s1[r] - mnew); psum += s1[r]; }
    psum += __shfl_xor(psum, 16);
    lsum = lsum * alpha + psum;

#pragma unroll
    for (int r = 0; r < 8; ++r) {
      y0[r] *= alpha; y1[r] *= alpha; y2[r] *= alpha; y3[r] *= alpha;
    }

    // --- P^T (D-layout) -> PV B-operand via half-wave swap ---
    v16bf B2;
#pragma unroll
    for (int r = 0; r < 8; ++r) {
      const float q0 = __shfl_xor(s0[r], 16);
      const float q1 = __shfl_xor(s1[r], 16);
      B2[r]     = f2bf(hl ? q1 : s0[r]);   // K = hl*16 + r
      B2[r + 8] = f2bf(hl ? s1[r] : q0);   // K = hl*16 + 8 + r
    }

    // --- y += g_tile(16o x 32m) * P(32m x 16q): 4 o-chunks sharing B2 ---
    const bf16_t* gr = gB + (size_t)ln * NN + m0;
    y0 = wmma_bf16   (ldA_bf(gr, hl), B2, y0);
    y1 = wmma_bf16_rb(ldA_bf(gr + (size_t)16 * NN, hl), B2, y1);
    y2 = wmma_bf16_rb(ldA_bf(gr + (size_t)32 * NN, hl), B2, y2);
    y3 = wmma_bf16_rb(ldA_bf(gr + (size_t)48 * NN, hl), B2, y3);
  }

  // --- normalize ---
  const float inv = 1.0f / lsum;
#pragma unroll
  for (int r = 0; r < 8; ++r) {
    y0[r] *= inv; y1[r] *= inv; y2[r] *= inv; y3[r] *= inv;
  }

  // --- y (D-layout) -> B-operands for the output projection ---
  v16bf B30, B31;
#pragma unroll
  for (int r = 0; r < 8; ++r) {
    const float a0 = y0[r], a1 = y1[r], a2 = y2[r], a3 = y3[r];
    const float t0 = __shfl_xor(a0, 16), t1 = __shfl_xor(a1, 16);
    const float t2 = __shfl_xor(a2, 16), t3 = __shfl_xor(a3, 16);
    B30[r]     = f2bf(hl ? t1 : a0);   // K = o 0..31
    B30[r + 8] = f2bf(hl ? a1 : t0);
    B31[r]     = f2bf(hl ? t3 : a2);   // K = o 32..63
    B31[r + 8] = f2bf(hl ? a3 : t2);
  }

  // --- z = relu(w_out*y + b_out) * (gamma*rsqrt(1+eps)) + beta; out = x+z ---
  // Two passes of 4 consecutive WMMAs so each pass shares its B operand.
  v8f z0 = {}, z1 = {}, z2 = {}, z3 = {};
  {
    const size_t r0 = (size_t)ln * CH;
    z0 = wmma_bf16   (ldA_bf(wob + r0, hl), B30, z0);
    z1 = wmma_bf16_rb(ldA_bf(wob + r0 + (size_t)16 * CH, hl), B30, z1);
    z2 = wmma_bf16_rb(ldA_bf(wob + r0 + (size_t)32 * CH, hl), B30, z2);
    z3 = wmma_bf16_rb(ldA_bf(wob + r0 + (size_t)48 * CH, hl), B30, z3);
    z0 = wmma_bf16   (ldA_bf(wob + r0 + 32, hl), B31, z0);
    z1 = wmma_bf16_rb(ldA_bf(wob + r0 + (size_t)16 * CH + 32, hl), B31, z1);
    z2 = wmma_bf16_rb(ldA_bf(wob + r0 + (size_t)32 * CH + 32, hl), B31, z2);
    z3 = wmma_bf16_rb(ldA_bf(wob + r0 + (size_t)48 * CH + 32, hl), B31, z3);
  }

  const float rs = rsqrtf(1.0f + 1e-5f);
#pragma unroll
  for (int i = 0; i < 4; ++i) {
    const v8f z = (i == 0) ? z0 : (i == 1) ? z1 : (i == 2) ? z2 : z3;
    const int cbase = 16 * i + 8 * hl;
#pragma unroll
    for (int r = 0; r < 8; ++r) {
      const int c = cbase + r;
      float zz = z[r] + b_out[c];
      zz = fmaxf(zz, 0.0f);
      zz = zz * (gamma[c] * rs) + beta[c];
      const size_t off = (size_t)(b * CH + c) * NN + n0 + ln;
      out[off] = x[off] + zz;
    }
  }
}

// ---------------------------------------------------------------------------
extern "C" void kernel_launch(void* const* d_in, const int* in_sizes, int n_in,
                              void* d_out, int out_size, void* d_ws,
                              size_t ws_size, hipStream_t stream) {
  const float* x       = (const float*)d_in[0];
  const float* w_theta = (const float*)d_in[1];
  const float* b_theta = (const float*)d_in[2];
  const float* w_phi   = (const float*)d_in[3];
  const float* b_phi   = (const float*)d_in[4];
  const float* w_g     = (const float*)d_in[5];
  const float* b_g     = (const float*)d_in[6];
  const float* w_out   = (const float*)d_in[7];
  const float* b_out   = (const float*)d_in[8];
  const float* gamma   = (const float*)d_in[9];
  const float* beta    = (const float*)d_in[10];

  bf16_t* thetaT = (bf16_t*)d_ws;                       // [B][N][C] bf16: 2 MB
  bf16_t* phiT   = thetaT + (size_t)NB * NN * CH;       // [B][N][C] bf16: 2 MB
  bf16_t* gbuf   = phiT + (size_t)NB * NN * CH;         // [B][C][N] bf16: 2 MB
  bf16_t* wb     = gbuf + (size_t)NB * CH * NN;         // 4 x [64][64] bf16
  bf16_t* wtb = wb, *wpb = wb + CH * CH, *wgb = wb + 2 * CH * CH,
        *wob = wb + 3 * CH * CH;

  wcvt_kernel<<<dim3(64), dim3(256), 0, stream>>>(w_theta, w_phi, w_g, w_out, wb);
  proj_kernel<<<dim3(128), dim3(256), 0, stream>>>(
      x, wtb, wpb, wgb, b_theta, b_phi, b_g, thetaT, phiT, gbuf);
  attn_kernel<<<dim3(128), dim3(256), 0, stream>>>(
      thetaT, phiT, gbuf, wob, b_out, gamma, beta, x, (float*)d_out);
}
